// NTM_87625922773399
// MI455X (gfx1250) — matrix operand order repaired and checked by reading, not compiled, not run on hardware
//
#include <hip/hip_runtime.h>
#include <hip/hip_bf16.h>
#include <stdint.h>

#define BB 64
#define TT 128
#define INDIM 264
#define HST 512
#define NH 4
#define MM 128
#define AA 2048
#define RDSZ 512
#define PPH 390
#define UPDN 1560
#define UPDP 1568      // padded to 98*16
#define CTRLK 1288
#define CTRLKP 1312    // padded to 41*32
#define KT_STATE 41
#define NT_STATE 32
#define KT_H 16
#define NT_OUT 16
#define NT_UPD 98
#define EPSV 1e-8f

typedef __bf16        v16bf __attribute__((ext_vector_type(16)));
typedef float         v8f   __attribute__((ext_vector_type(8)));
typedef unsigned int  v8u   __attribute__((ext_vector_type(8)));
typedef unsigned int  v4u   __attribute__((ext_vector_type(4)));

__device__ inline unsigned short f32_to_bf16(float f) {
  union { float f; unsigned int u; } x; x.f = f;
  unsigned int r = x.u + 0x7FFFu + ((x.u >> 16) & 1u);
  return (unsigned short)(r >> 16);
}
__device__ inline float sigm(float x) { return 1.f / (1.f + __expf(-x)); }
__device__ inline float softpl(float x) { return (x > 20.f) ? x : log1pf(__expf(x)); }

// ---------------- init: mem=0.01, wt one-hot, h=1 ----------------
__global__ void k_init(float* mem, float* wt, float* h) {
  size_t i = (size_t)blockIdx.x * blockDim.x + threadIdx.x;
  if (i < (size_t)BB * MM * AA) mem[i] = 0.01f;
  if (i < (size_t)BB * NH * AA) wt[i] = ((i % AA) == 0) ? 1.0f : 0.0f;
  if (i < (size_t)BB * HST)     h[i]  = 1.0f;
}

// ------- pack row-major W[K][N] f32 -> bf16 WMMA B-fragments -------
// layout: frag = nt*KT + kt ; out[frag*256 + lane*8 + v]
// lane<16:  n = nt*16+lane,    rows K = kt*32 + {2v, 2v+1}
// lane>=16: n = nt*16+lane-16, rows K = kt*32 + 16 + {2v, 2v+1}
__global__ void k_pack(const float* __restrict__ W, unsigned int* __restrict__ out,
                       int KT, int NT, int Kreal, int Nreal) {
  int i = blockIdx.x * blockDim.x + threadIdx.x;
  if (i >= NT * KT * 256) return;
  int v = i & 7, lane = (i >> 3) & 31, frag = i >> 8;
  int kt = frag % KT, nt = frag / KT;
  int n  = nt * 16 + (lane & 15);
  int k0 = kt * 32 + ((lane >> 4) << 4) + 2 * v;
  float lo = (k0     < Kreal && n < Nreal) ? W[(size_t)k0       * Nreal + n] : 0.f;
  float hi = (k0 + 1 < Kreal && n < Nreal) ? W[(size_t)(k0 + 1) * Nreal + n] : 0.f;
  out[i] = (unsigned int)f32_to_bf16(lo) | ((unsigned int)f32_to_bf16(hi) << 16);
}

// ---------------- read[b,h,m] = sum_a mem[b,m,a] * wt[b,h,a] ----------------
// wt[b] staged once in LDS (32 KB) to avoid 128x redundant global re-reads
__global__ void __launch_bounds__(256)
k_read(const float* __restrict__ mem, const float* __restrict__ wt,
       float* __restrict__ rd) {
  int b = blockIdx.x;
  int tid = threadIdx.x;
  int wave = tid >> 5, lane = tid & 31;
  __shared__ float wtL[NH][AA];                       // 32 KB
  for (int i = tid; i < NH * AA; i += 256)
    wtL[i >> 11][i & (AA - 1)] = wt[(size_t)b * NH * AA + i];
  __syncthreads();
  for (int m = wave; m < MM; m += 8) {
    const float* row = mem + ((size_t)b * MM + m) * AA;
    float acc0 = 0.f, acc1 = 0.f, acc2 = 0.f, acc3 = 0.f;
    for (int a = lane; a < AA; a += 32) {
      float mv = row[a];
      acc0 = fmaf(mv, wtL[0][a], acc0);
      acc1 = fmaf(mv, wtL[1][a], acc1);
      acc2 = fmaf(mv, wtL[2][a], acc2);
      acc3 = fmaf(mv, wtL[3][a], acc3);
    }
    for (int off = 16; off > 0; off >>= 1) {
      acc0 += __shfl_xor(acc0, off, 32); acc1 += __shfl_xor(acc1, off, 32);
      acc2 += __shfl_xor(acc2, off, 32); acc3 += __shfl_xor(acc3, off, 32);
    }
    if (lane == 0) {
      rd[(size_t)b * RDSZ + 0 * MM + m] = acc0;
      rd[(size_t)b * RDSZ + 1 * MM + m] = acc1;
      rd[(size_t)b * RDSZ + 2 * MM + m] = acc2;
      rd[(size_t)b * RDSZ + 3 * MM + m] = acc3;
    }
  }
}

// ---------------- assemble ctrl = bf16([x_t | h | read | 0-pad]) ----------------
__global__ void k_ctrl(const float* __restrict__ x_t, int x_stride,
                       const float* __restrict__ h, const float* __restrict__ rd,
                       unsigned short* __restrict__ ctrl) {
  int i = blockIdx.x * blockDim.x + threadIdx.x;
  if (i >= BB * CTRLKP) return;
  int b = i / CTRLKP, c = i % CTRLKP;
  float v = 0.f;
  if (c < INDIM)            v = x_t[(size_t)b * x_stride + c];
  else if (c < INDIM + HST) v = h[(size_t)b * HST + (c - INDIM)];
  else if (c < CTRLK)       v = rd[(size_t)b * RDSZ + (c - INDIM - HST)];
  ctrl[i] = f32_to_bf16(v);
}

// ---------------- skinny GEMM: Y[64,N] = act(A @ W + b), WMMA bf16 ----------------
// one wave per 16x16 output tile; dual accumulators break the WMMA->WMMA
// RAW chain (5-slot bf16 hazard, ISA 7.12.1) so the scheduler overlaps instead
// of inserting v_nops.  A row-major bf16 (lda elems, 16B-aligned rows).
// mode 0: sigmoid -> outF (ldo) + bf16 copy to outBf(stride 512)
// mode 1: sigmoid -> outF ; mode 2: linear -> outF (guard n < Nreal)
__global__ void k_gemm(const unsigned short* __restrict__ Abf, int lda,
                       const unsigned int* __restrict__ Wpk,
                       const float* __restrict__ bias,
                       int KT, int NT, int Nreal, int mode,
                       float* __restrict__ outF, int ldo,
                       unsigned short* __restrict__ outBf) {
  int wave = (blockIdx.x * blockDim.x + threadIdx.x) >> 5;
  int lane = threadIdx.x & 31;
  int mtile = wave & 3;
  int nt = wave >> 2;
  if (nt >= NT) return;

  int row = mtile * 16 + (lane & 15);
  int ko  = (lane >> 4) * 16;                // K sub-offset per lane half
  const unsigned short* arow = Abf + (size_t)row * lda;
  const unsigned int*   wbase = Wpk + (((size_t)nt * KT) << 8) + lane * 8;

  v8f acc0 = {}, acc1 = {};
  int kt = 0;
  for (; kt + 1 < KT; kt += 2) {
    union { v4u q[2]; v16bf v; } a0, a1;
    a0.q[0] = *(const v4u*)(arow + kt * 32 + ko);
    a0.q[1] = *(const v4u*)(arow + kt * 32 + 8 + ko);
    a1.q[0] = *(const v4u*)(arow + kt * 32 + 32 + ko);
    a1.q[1] = *(const v4u*)(arow + kt * 32 + 40 + ko);
    union { v8u u; v16bf v; } b0, b1;
    b0.u = *(const v8u*)(wbase + ((size_t)kt << 8));
    b1.u = *(const v8u*)(wbase + ((size_t)(kt + 1) << 8));
    acc0 = __builtin_amdgcn_wmma_f32_16x16x32_bf16(false, a0.v, false, b0.v,
                                                   (short)0, acc0, false, false);
    acc1 = __builtin_amdgcn_wmma_f32_16x16x32_bf16(false, a1.v, false, b1.v,
                                                   (short)0, acc1, false, false);
  }
  if (kt < KT) {
    union { v4u q[2]; v16bf v; } a;
    a.q[0] = *(const v4u*)(arow + kt * 32 + ko);
    a.q[1] = *(const v4u*)(arow + kt * 32 + 8 + ko);
    union { v8u u; v16bf v; } bm;
    bm.u = *(const v8u*)(wbase + ((size_t)kt << 8));
    acc0 = __builtin_amdgcn_wmma_f32_16x16x32_bf16(false, a.v, false, bm.v,
                                                   (short)0, acc0, false, false);
  }
  v8f acc = acc0 + acc1;

  int n = nt * 16 + (lane & 15);
  int mbase = mtile * 16 + ((lane >> 4) << 3);
  float bv = (n < Nreal) ? bias[n] : 0.f;
#pragma unroll
  for (int r = 0; r < 8; ++r) {
    int m = mbase + r;
    float val = acc[r] + bv;
    if (mode == 0) {
      float s = sigm(val);
      outF[(size_t)m * ldo + n] = s;
      outBf[(size_t)m * HST + n] = f32_to_bf16(s);
    } else if (mode == 1) {
      outF[(size_t)m * ldo + n] = sigm(val);
    } else {
      if (n < Nreal) outF[(size_t)m * ldo + n] = val;
    }
  }
}

// ---------------- split upd row into activated head params ----------------
__global__ void k_updact(const float* __restrict__ upd,
                         float* __restrict__ kvec, float* __restrict__ ers,
                         float* __restrict__ adv, float* __restrict__ knorm,
                         float* __restrict__ beta, float* __restrict__ gv,
                         float* __restrict__ gmv, float* __restrict__ shv) {
  int bh = blockIdx.x; int b = bh >> 2, hh = bh & 3;
  const float* u = upd + (size_t)b * UPDP + hh * PPH;
  int t = threadIdx.x;                       // 128 threads
  __shared__ float red[128];
  float kv = u[t];
  kvec[(size_t)bh * MM + t] = kv;
  red[t] = kv * kv;
  __syncthreads();
  for (int s = 64; s > 0; s >>= 1) { if (t < s) red[t] += red[t + s]; __syncthreads(); }
  if (t == 0) {
    knorm[bh] = sqrtf(red[0]);
    beta[bh]  = softpl(u[MM]);
    gv[bh]    = sigm(u[MM + 1]);
    float s0 = u[MM + 2], s1 = u[MM + 3], s2 = u[MM + 4];
    float mx = fmaxf(s0, fmaxf(s1, s2));
    float e0 = __expf(s0 - mx), e1 = __expf(s1 - mx), e2 = __expf(s2 - mx);
    float si = 1.f / (e0 + e1 + e2);
    shv[bh * 3 + 0] = e0 * si; shv[bh * 3 + 1] = e1 * si; shv[bh * 3 + 2] = e2 * si;
    gmv[bh] = 1.f + softpl(u[MM + 5]);
  }
  ers[(size_t)bh * MM + t] = sigm(u[MM + 6 + t]);
  adv[(size_t)bh * MM + t] = tanhf(u[MM + 6 + MM + t]);
}

// ------- addressing: dots, cosine sim, softmax, gate, shift, sharpen -> wt -------
__global__ void __launch_bounds__(256)
k_addr(const float* __restrict__ mem, float* __restrict__ wt,
       const float* __restrict__ kvec, const float* __restrict__ knorm,
       const float* __restrict__ beta, const float* __restrict__ gv,
       const float* __restrict__ gmv, const float* __restrict__ shv) {
  int b = blockIdx.x, t = threadIdx.x;       // 256 threads, 8 a's each
  __shared__ float kvL[NH][MM];              // 2 KB
  __shared__ float wg[NH][AA];               // 32 KB
  __shared__ float red[256];
  for (int i = t; i < NH * MM; i += 256) kvL[i >> 7][i & 127] = kvec[(size_t)b * NH * MM + i];
  __syncthreads();

  float acc[NH][8], msq[8];
#pragma unroll
  for (int j = 0; j < 8; ++j) { msq[j] = 0.f;
#pragma unroll
    for (int h = 0; h < NH; ++h) acc[h][j] = 0.f; }

  for (int m = 0; m < MM; ++m) {
    const float* row = mem + ((size_t)b * MM + m) * AA;
#pragma unroll
    for (int j = 0; j < 8; ++j) {
      float mv = row[t + j * 256];
      msq[j] = fmaf(mv, mv, msq[j]);
#pragma unroll
      for (int h = 0; h < NH; ++h) acc[h][j] = fmaf(kvL[h][m], mv, acc[h][j]);
    }
  }

  for (int h = 0; h < NH; ++h) {
    float kn = knorm[b * NH + h], bt = beta[b * NH + h], gg = gv[b * NH + h];
    float e[8], lm = -1e30f;
#pragma unroll
    for (int j = 0; j < 8; ++j) {
      float sim = acc[h][j] / (kn * sqrtf(msq[j]) + EPSV);
      e[j] = bt * sim; lm = fmaxf(lm, e[j]);
    }
    red[t] = lm; __syncthreads();
    for (int s = 128; s > 0; s >>= 1) { if (t < s) red[t] = fmaxf(red[t], red[t + s]); __syncthreads(); }
    float mx = red[0]; __syncthreads();
    float ls = 0.f;
#pragma unroll
    for (int j = 0; j < 8; ++j) { e[j] = __expf(e[j] - mx); ls += e[j]; }
    red[t] = ls; __syncthreads();
    for (int s = 128; s > 0; s >>= 1) { if (t < s) red[t] += red[t + s]; __syncthreads(); }
    float inv = 1.f / red[0]; __syncthreads();
#pragma unroll
    for (int j = 0; j < 8; ++j) {
      int a = t + j * 256;
      float wold = wt[((size_t)b * NH + h) * AA + a];
      wg[h][a] = gg * (e[j] * inv) + (1.f - gg) * wold;
    }
    __syncthreads();
  }

  for (int h = 0; h < NH; ++h) {
    float s0 = shv[(b * NH + h) * 3 + 0];
    float s1 = shv[(b * NH + h) * 3 + 1];
    float s2 = shv[(b * NH + h) * 3 + 2];
    float gm = gmv[b * NH + h];
    float wp[8], ls = 0.f;
#pragma unroll
    for (int j = 0; j < 8; ++j) {
      int a = t + j * 256;
      float ws = s0 * wg[h][(a + 1) & (AA - 1)] + s1 * wg[h][a] + s2 * wg[h][(a - 1) & (AA - 1)];
      wp[j] = __powf(ws + EPSV, gm);
      ls += wp[j];
    }
    red[t] = ls; __syncthreads();
    for (int s = 128; s > 0; s >>= 1) { if (t < s) red[t] += red[t + s]; __syncthreads(); }
    float inv = 1.f / red[0]; __syncthreads();
#pragma unroll
    for (int j = 0; j < 8; ++j)
      wt[((size_t)b * NH + h) * AA + (t + j * 256)] = wp[j] * inv;
  }
}

// ---------------- mem = mem * prod_h(1 - e_h w_h) + sum_h a_h w_h ----------------
__global__ void k_memupd(float* __restrict__ mem, const float* __restrict__ wt,
                         const float* __restrict__ ers, const float* __restrict__ adv) {
  int bm = blockIdx.x; int b = bm >> 7, m = bm & 127;
  int t = threadIdx.x;
  float er[NH], ad[NH];
#pragma unroll
  for (int h = 0; h < NH; ++h) {
    er[h] = ers[((size_t)b * NH + h) * MM + m];
    ad[h] = adv[((size_t)b * NH + h) * MM + m];
  }
  float* row = mem + ((size_t)b * MM + m) * AA;
#pragma unroll
  for (int j = 0; j < 8; ++j) {
    int a = t + j * 256;
    float ep = 1.f, at = 0.f;
#pragma unroll
    for (int h = 0; h < NH; ++h) {
      float w = wt[((size_t)b * NH + h) * AA + a];
      ep *= (1.f - er[h] * w);
      at = fmaf(ad[h], w, at);
    }
    row[a] = row[a] * ep + at;
  }
}

extern "C" void kernel_launch(void* const* d_in, const int* in_sizes, int n_in,
                              void* d_out, int out_size, void* d_ws, size_t ws_size,
                              hipStream_t stream) {
  (void)in_sizes; (void)n_in; (void)out_size; (void)ws_size;
  const float* x   = (const float*)d_in[0];
  const float* Wst = (const float*)d_in[1];
  const float* bst = (const float*)d_in[2];
  const float* Wo  = (const float*)d_in[3];
  const float* bo  = (const float*)d_in[4];
  const float* Wu  = (const float*)d_in[5];
  const float* bu  = (const float*)d_in[6];
  float* out = (float*)d_out;

  char* p = (char*)d_ws;
  auto alloc = [&](size_t bytes) -> void* {
    void* r = (void*)p; p += (bytes + 255) & ~(size_t)255; return r;
  };
  float*          mem  = (float*)alloc((size_t)BB * MM * AA * 4);
  float*          wt   = (float*)alloc((size_t)BB * NH * AA * 4);
  float*          h    = (float*)alloc((size_t)BB * HST * 4);
  unsigned short* hbf  = (unsigned short*)alloc((size_t)BB * HST * 2);
  float*          rd   = (float*)alloc((size_t)BB * RDSZ * 4);
  unsigned short* ctrl = (unsigned short*)alloc((size_t)BB * CTRLKP * 2);
  float*          updb = (float*)alloc((size_t)BB * UPDP * 4);
  float*          kvec = (float*)alloc((size_t)BB * NH * MM * 4);
  float*          ers  = (float*)alloc((size_t)BB * NH * MM * 4);
  float*          adv  = (float*)alloc((size_t)BB * NH * MM * 4);
  float*          knm  = (float*)alloc((size_t)BB * NH * 4);
  float*          bet  = (float*)alloc((size_t)BB * NH * 4);
  float*          gv   = (float*)alloc((size_t)BB * NH * 4);
  float*          gmv  = (float*)alloc((size_t)BB * NH * 4);
  float*          shv  = (float*)alloc((size_t)BB * NH * 3 * 4);
  unsigned int*   wpkS = (unsigned int*)alloc((size_t)NT_STATE * KT_STATE * 256 * 4);
  unsigned int*   wpkO = (unsigned int*)alloc((size_t)NT_OUT   * KT_H     * 256 * 4);
  unsigned int*   wpkU = (unsigned int*)alloc((size_t)NT_UPD   * KT_H     * 256 * 4);

  // one-time prep (per launch; deterministic)
  {
    size_t n = (size_t)BB * MM * AA;
    k_init<<<(unsigned)((n + 255) / 256), 256, 0, stream>>>(mem, wt, h);
    k_pack<<<(NT_STATE * KT_STATE * 256 + 255) / 256, 256, 0, stream>>>(Wst, wpkS, KT_STATE, NT_STATE, CTRLK, HST);
    k_pack<<<(NT_OUT * KT_H * 256 + 255) / 256, 256, 0, stream>>>(Wo, wpkO, KT_H, NT_OUT, HST, 256);
    k_pack<<<(NT_UPD * KT_H * 256 + 255) / 256, 256, 0, stream>>>(Wu, wpkU, KT_H, NT_UPD, HST, UPDN);
  }

  for (int t = 0; t < TT; ++t) {
    k_read<<<BB, 256, 0, stream>>>(mem, wt, rd);
    k_ctrl<<<(BB * CTRLKP + 255) / 256, 256, 0, stream>>>(x + (size_t)t * INDIM, TT * INDIM, h, rd, ctrl);
    k_gemm<<<(4 * NT_STATE + 7) / 8, 256, 0, stream>>>(ctrl, CTRLKP, wpkS, bst, KT_STATE, NT_STATE, HST, 0, h, HST, hbf);
    k_gemm<<<(4 * NT_OUT + 7) / 8, 256, 0, stream>>>(hbf, HST, wpkO, bo, KT_H, NT_OUT, 256, 1, out + (size_t)t * 256, TT * 256, (unsigned short*)nullptr);
    k_gemm<<<(4 * NT_UPD + 7) / 8, 256, 0, stream>>>(hbf, HST, wpkU, bu, KT_H, NT_UPD, UPDN, 2, updb, UPDP, (unsigned short*)nullptr);
    k_updact<<<BB * NH, 128, 0, stream>>>(updb, kvec, ers, adv, knm, bet, gv, gmv, shv);
    k_addr<<<BB, 256, 0, stream>>>(mem, wt, kvec, knm, bet, gv, gmv, shv);
    k_memupd<<<BB * MM, 256, 0, stream>>>(mem, wt, ers, adv);
  }
}
